// BidirectionalMambaLayer_8254927143299
// MI455X (gfx1250) — matrix operand order repaired and checked by reading, not compiled, not run on hardware
//
#include <hip/hip_runtime.h>
#include <hip/hip_bf16.h>
#include <math.h>

#define D_MODEL   512
#define D_STATE   128
#define D_CONV    4
#define D_INNER   1024
#define NHEADS    16
#define HEADDIM   64
#define D_XBC     1280
#define D_IN_PROJ 2320
#define B_SZ      4
#define SEQ       2048
#define NTOK      (B_SZ*SEQ)   /* 8192 */
#define EPS       1e-5f

typedef __bf16 bf16_t;
typedef __attribute__((ext_vector_type(16))) __bf16 v16bf;
typedef __attribute__((ext_vector_type(8)))  float  v8f;

// ---------------- elementwise conversion / transpose ----------------

__global__ __launch_bounds__(256) void k_convert_bf16(const float* __restrict__ in,
                                                      bf16_t* __restrict__ out, int n) {
    int i = blockIdx.x*256 + threadIdx.x;
    if (i < n) out[i] = (bf16_t)in[i];
}

// in: [K][N] f32 row-major  ->  out: [N][K] bf16 row-major
__global__ __launch_bounds__(256) void k_transpose_bf16(const float* __restrict__ in,
                                                        bf16_t* __restrict__ out, int K, int N) {
    int i = blockIdx.x*256 + threadIdx.x;
    if (i < K*N) {
        int k = i / N, n = i % N;
        out[(size_t)n*K + k] = (bf16_t)in[i];
    }
}

// ---------------- bf16 WMMA GEMM: C[M][N] = A[M][K] * Bt[N][K]^T ----------------
// mode 0: store C
// mode 1: C[row*N+col]  = mask * acc   (masked store to d_out)
// mode 2: C[row*N+col] += mask * acc   (masked accumulate into d_out)
// Requires: M % 64 == 0, K % 32 == 0. N guarded.

union FragU { uint4 q[2]; v16bf v; };

__global__ __launch_bounds__(256) void k_gemm_bf16_wmma(
    const bf16_t* __restrict__ A, const bf16_t* __restrict__ Bt,
    float* __restrict__ C, int M, int N, int K, int mode,
    const unsigned char* __restrict__ pad)
{
    const int tid  = threadIdx.x;
    const int lane = tid & 31;
    const int w    = tid >> 5;         // 8 waves
    const int wm   = w >> 2;           // 0..1  (M tiles of 32)
    const int wn   = w & 3;            // 0..3  (N tiles of 32)
    const int half = lane >> 4;        // 0/1
    const int lm   = lane & 15;

    const int row0 = blockIdx.y*64  + wm*32;
    const int col0 = blockIdx.x*128 + wn*32;

    v8f acc00 = {}, acc01 = {}, acc10 = {}, acc11 = {};

    // A fragment rows (16-bit A 16x32 layout: elems 0..7 = K[half*8 .. +8),
    //                                          elems 8..15 = K[16+half*8 .. +8))
    const bf16_t* pa0 = A + (size_t)(row0 + lm)      * K + half*8;
    const bf16_t* pa1 = A + (size_t)(row0 + 16 + lm) * K + half*8;
    // B fragment cols (B 32x16 layout: elems 0..7 = K[half*8 .. +8), elems 8..15 = K[16+half*8 .. +8), N = lm)
    const int n0 = col0 + lm, n1 = col0 + 16 + lm;
    const bf16_t* pb0 = Bt + (size_t)((n0 < N) ? n0 : 0) * K + half*8;
    const bf16_t* pb1 = Bt + (size_t)((n1 < N) ? n1 : 0) * K + half*8;

    for (int kb = 0; kb < K; kb += 32) {
        FragU a0, a1, b0, b1;
        a0.q[0] = *(const uint4*)(pa0 + kb);
        a0.q[1] = *(const uint4*)(pa0 + kb + 16);
        a1.q[0] = *(const uint4*)(pa1 + kb);
        a1.q[1] = *(const uint4*)(pa1 + kb + 16);
        b0.q[0] = *(const uint4*)(pb0 + kb);
        b0.q[1] = *(const uint4*)(pb0 + kb + 16);
        b1.q[0] = *(const uint4*)(pb1 + kb);
        b1.q[1] = *(const uint4*)(pb1 + kb + 16);
        acc00 = __builtin_amdgcn_wmma_f32_16x16x32_bf16(false, a0.v, false, b0.v, (short)0, acc00, false, false);
        acc01 = __builtin_amdgcn_wmma_f32_16x16x32_bf16(false, a0.v, false, b1.v, (short)0, acc01, false, false);
        acc10 = __builtin_amdgcn_wmma_f32_16x16x32_bf16(false, a1.v, false, b0.v, (short)0, acc10, false, false);
        acc11 = __builtin_amdgcn_wmma_f32_16x16x32_bf16(false, a1.v, false, b1.v, (short)0, acc11, false, false);
    }

    // C/D layout: VGPR r -> M = r + half*8, N = lm (within 16x16 tile)
    auto store_tile = [&](const v8f& acc, int trow, int tcol) {
        int col = tcol + lm;
        if (col >= N) return;
        #pragma unroll
        for (int r = 0; r < 8; ++r) {
            int row = trow + half*8 + r;
            float v = acc[r];
            size_t o = (size_t)row*N + col;
            if (mode == 0) {
                C[o] = v;
            } else {
                float m = pad[row] ? 0.0f : 1.0f;
                if (mode == 1) C[o] = m*v;
                else           C[o] += m*v;
            }
        }
    };
    store_tile(acc00, row0,      col0);
    store_tile(acc01, row0,      col0+16);
    store_tile(acc10, row0+16,   col0);
    store_tile(acc11, row0+16,   col0+16);
}

// ---------------- causal conv (dir-aware) + SiLU ----------------

__global__ __launch_bounds__(256) void k_conv_silu(
    const float* __restrict__ zx, const float* __restrict__ convw,
    const float* __restrict__ convb, float* __restrict__ xbc, int dir)
{
    int i = blockIdx.x*256 + threadIdx.x;
    if (i >= NTOK * D_XBC) return;
    int c   = i % D_XBC;
    int tok = i / D_XBC;
    int b = tok / SEQ, t = tok % SEQ;
    float acc = convb[c];
    #pragma unroll
    for (int k = 0; k < D_CONV; ++k) {
        int tt = dir ? (t + (D_CONV-1) - k) : (t - (D_CONV-1) + k);
        if (tt >= 0 && tt < SEQ)
            acc = fmaf(convw[k*D_XBC + c],
                       zx[(size_t)(b*SEQ + tt)*D_IN_PROJ + D_INNER + c], acc);
    }
    xbc[(size_t)tok*D_XBC + c] = acc / (1.0f + expf(-acc));   // silu
}

// ---------------- dt = softplus(raw + bias), dA = exp(-exp(A_log)*dt) ----------------

__global__ __launch_bounds__(256) void k_dt_dA(
    const float* __restrict__ zx, const float* __restrict__ dt_bias,
    const float* __restrict__ A_log, float* __restrict__ dt, float* __restrict__ dA)
{
    int i = blockIdx.x*256 + threadIdx.x;
    if (i >= NTOK * NHEADS) return;
    int h = i % NHEADS, tok = i / NHEADS;
    float raw = zx[(size_t)tok*D_IN_PROJ + D_INNER + D_XBC + h] + dt_bias[h];
    float d = (raw > 20.0f) ? raw : log1pf(expf(raw));
    dt[i] = d;
    dA[i] = expf(-expf(A_log[h]) * d);
}

// ---------------- selective scan ----------------
// grid: (B*NHEADS, 2 state-splits of 64, 2 directions); block 256.
// thread owns p = tid>>2 and 16 n-values (qq = tid&3); reduce over 4 lanes with shfl.

#define CHUNK 16
__global__ __launch_bounds__(256) void k_scan(
    const float* __restrict__ xbc_f, const float* __restrict__ xbc_b,
    const float* __restrict__ dt_f,  const float* __restrict__ dt_b,
    const float* __restrict__ dA_f,  const float* __restrict__ dA_b,
    float* __restrict__ y_f, float* __restrict__ y_b)
{
    const int bh = blockIdx.x;
    const int b = bh / NHEADS, hh = bh % NHEADS;
    const int nbase = blockIdx.y * 64;
    const int dir = blockIdx.z;
    const float* xbc = dir ? xbc_b : xbc_f;
    const float* dtp = dir ? dt_b  : dt_f;
    const float* dAp = dir ? dA_b  : dA_f;
    float* y         = dir ? y_b   : y_f;

    const int tid = threadIdx.x;
    const int p = tid >> 2, qq = tid & 3;

    __shared__ float sX[CHUNK][64];
    __shared__ float sB[CHUNK][64];
    __shared__ float sC[CHUNK][64];
    __shared__ float sdt[CHUNK], sdA[CHUNK];

    float h[16];
    #pragma unroll
    for (int j = 0; j < 16; ++j) h[j] = 0.0f;

    for (int ch = 0; ch < SEQ/CHUNK; ++ch) {
        for (int u = tid; u < CHUNK*64; u += 256) {
            int i = u >> 6, ln = u & 63;
            int s = ch*CHUNK + i;
            int t = dir ? (SEQ-1 - s) : s;
            size_t row = (size_t)(b*SEQ + t)*D_XBC;
            sX[i][ln] = xbc[row + hh*HEADDIM + ln];
            sB[i][ln] = xbc[row + D_INNER + nbase + ln];
            sC[i][ln] = xbc[row + D_INNER + D_STATE + nbase + ln];
        }
        if (tid < CHUNK) {
            int s = ch*CHUNK + tid;
            int t = dir ? (SEQ-1 - s) : s;
            sdt[tid] = dtp[(size_t)(b*SEQ + t)*NHEADS + hh];
            sdA[tid] = dAp[(size_t)(b*SEQ + t)*NHEADS + hh];
        }
        __syncthreads();
        #pragma unroll 4
        for (int i = 0; i < CHUNK; ++i) {
            float dAi = sdA[i];
            float xp  = sX[i][p] * sdt[i];
            float cacc = 0.0f;
            const float* Brow = &sB[i][qq*16];
            const float* Crow = &sC[i][qq*16];
            #pragma unroll
            for (int j = 0; j < 16; ++j) {
                h[j] = fmaf(h[j], dAi, xp * Brow[j]);
                cacc = fmaf(h[j], Crow[j], cacc);
            }
            cacc += __shfl_xor(cacc, 1, 32);
            cacc += __shfl_xor(cacc, 2, 32);
            if (qq == 0) {
                int s = ch*CHUNK + i;
                int t = dir ? (SEQ-1 - s) : s;
                atomicAdd(&y[(size_t)(b*SEQ + t)*D_INNER + hh*HEADDIM + p], cacc);
            }
        }
        __syncthreads();
    }
}

// ---------------- gate (silu(z)) + D*x skip + RMSNorm, emit bf16 for out_proj ----------------

__global__ __launch_bounds__(256) void k_gate_norm(
    const float* __restrict__ y, const float* __restrict__ zx,
    const float* __restrict__ xbc, const float* __restrict__ Dp,
    const float* __restrict__ norm_w, bf16_t* __restrict__ ybf)
{
    const int row = blockIdx.x;
    const int tid = threadIdx.x;
    float v[4]; float ss = 0.0f;
    #pragma unroll
    for (int r = 0; r < 4; ++r) {
        int i = r*256 + tid;
        float z  = zx[(size_t)row*D_IN_PROJ + i];
        float yv = y[(size_t)row*D_INNER + i] + Dp[i >> 6] * xbc[(size_t)row*D_XBC + i];
        float g  = yv * (z / (1.0f + expf(-z)));
        v[r] = g; ss += g*g;
    }
    #pragma unroll
    for (int o = 16; o; o >>= 1) ss += __shfl_xor(ss, o, 32);
    __shared__ float red[8];
    if ((tid & 31) == 0) red[tid >> 5] = ss;
    __syncthreads();
    float tot = 0.0f;
    #pragma unroll
    for (int k = 0; k < 8; ++k) tot += red[k];
    float scale = rsqrtf(tot / (float)D_INNER + EPS);
    #pragma unroll
    for (int r = 0; r < 4; ++r) {
        int i = r*256 + tid;
        ybf[(size_t)row*D_INNER + i] = (bf16_t)(v[r] * scale * norm_w[i]);
    }
}

// ---------------- host orchestration ----------------

extern "C" void kernel_launch(void* const* d_in, const int* in_sizes, int n_in,
                              void* d_out, int out_size, void* d_ws, size_t ws_size,
                              hipStream_t stream)
{
    (void)in_sizes; (void)n_in; (void)out_size; (void)ws_size;

    const float* x            = (const float*)d_in[0];
    const unsigned char* pad  = (const unsigned char*)d_in[1];
    const float* w_in[2]   = { (const float*)d_in[2],  (const float*)d_in[10] };
    const float* conv_w[2] = { (const float*)d_in[3],  (const float*)d_in[11] };
    const float* conv_b[2] = { (const float*)d_in[4],  (const float*)d_in[12] };
    const float* dtb[2]    = { (const float*)d_in[5],  (const float*)d_in[13] };
    const float* alog[2]   = { (const float*)d_in[6],  (const float*)d_in[14] };
    const float* Dp[2]     = { (const float*)d_in[7],  (const float*)d_in[15] };
    const float* nw[2]     = { (const float*)d_in[8],  (const float*)d_in[16] };
    const float* w_out[2]  = { (const float*)d_in[9],  (const float*)d_in[17] };

    char* ws = (char*)d_ws;
    size_t off = 0;
    auto alloc = [&](size_t bytes) -> char* {
        char* p = ws + off;
        off += (bytes + 255) & ~(size_t)255;
        return p;
    };

    bf16_t* x_bf = (bf16_t*)alloc((size_t)NTOK*D_MODEL*sizeof(bf16_t));
    bf16_t *win_t[2], *wout_t[2], *ybf[2];
    float  *zx[2], *xbc[2], *dtv[2], *dAv[2], *yv[2];
    for (int d = 0; d < 2; ++d) {
        win_t[d]  = (bf16_t*)alloc((size_t)D_IN_PROJ*D_MODEL*sizeof(bf16_t));
        wout_t[d] = (bf16_t*)alloc((size_t)D_MODEL*D_INNER*sizeof(bf16_t));
        zx[d]  = (float*)alloc((size_t)NTOK*D_IN_PROJ*sizeof(float));
        xbc[d] = (float*)alloc((size_t)NTOK*D_XBC*sizeof(float));
        dtv[d] = (float*)alloc((size_t)NTOK*NHEADS*sizeof(float));
        dAv[d] = (float*)alloc((size_t)NTOK*NHEADS*sizeof(float));
        yv[d]  = (float*)alloc((size_t)NTOK*D_INNER*sizeof(float));
        ybf[d] = (bf16_t*)alloc((size_t)NTOK*D_INNER*sizeof(bf16_t));
    }

    // scan accumulates with atomics -> must zero y each call
    hipMemsetAsync(yv[0], 0, (size_t)NTOK*D_INNER*sizeof(float), stream);
    hipMemsetAsync(yv[1], 0, (size_t)NTOK*D_INNER*sizeof(float), stream);

    k_convert_bf16<<<(NTOK*D_MODEL+255)/256, 256, 0, stream>>>(x, x_bf, NTOK*D_MODEL);
    for (int d = 0; d < 2; ++d) {
        k_transpose_bf16<<<(D_MODEL*D_IN_PROJ+255)/256, 256, 0, stream>>>(w_in[d],  win_t[d],  D_MODEL, D_IN_PROJ);
        k_transpose_bf16<<<(D_INNER*D_MODEL+255)/256, 256, 0, stream>>>(w_out[d], wout_t[d], D_INNER, D_MODEL);
    }

    dim3 gin((D_IN_PROJ + 127)/128, NTOK/64);
    for (int d = 0; d < 2; ++d)
        k_gemm_bf16_wmma<<<gin, 256, 0, stream>>>(x_bf, win_t[d], zx[d],
                                                  NTOK, D_IN_PROJ, D_MODEL, 0, nullptr);

    for (int d = 0; d < 2; ++d) {
        k_conv_silu<<<(NTOK*D_XBC+255)/256, 256, 0, stream>>>(zx[d], conv_w[d], conv_b[d], xbc[d], d);
        k_dt_dA<<<(NTOK*NHEADS+255)/256, 256, 0, stream>>>(zx[d], dtb[d], alog[d], dtv[d], dAv[d]);
    }

    dim3 gscan(B_SZ*NHEADS, 2, 2);
    k_scan<<<gscan, 256, 0, stream>>>(xbc[0], xbc[1], dtv[0], dtv[1], dAv[0], dAv[1], yv[0], yv[1]);

    for (int d = 0; d < 2; ++d)
        k_gate_norm<<<NTOK, 256, 0, stream>>>(yv[d], zx[d], xbc[d], Dp[d], nw[d], ybf[d]);

    dim3 gout((D_MODEL + 127)/128, NTOK/64);
    k_gemm_bf16_wmma<<<gout, 256, 0, stream>>>(ybf[0], wout_t[0], (float*)d_out,
                                               NTOK, D_MODEL, D_INNER, 1, pad);
    k_gemm_bf16_wmma<<<gout, 256, 0, stream>>>(ybf[1], wout_t[1], (float*)d_out,
                                               NTOK, D_MODEL, D_INNER, 2, pad);
}
